// MultiHeadAttention_55027120996709
// MI455X (gfx1250) — matrix operand order repaired and checked by reading, not compiled, not run on hardware
//
#include <hip/hip_runtime.h>

typedef __attribute__((ext_vector_type(16))) _Float16 v16h;
typedef __attribute__((ext_vector_type(8)))  _Float16 v8h;
typedef __attribute__((ext_vector_type(8)))  float    v8f;

#define DMODEL 768
#define SLEN   512
#define BATCH  16
#define NHEAD  12
#define DK     64
#define NEG_BIG (-1.0e9f)

// A-matrix (16-bit, 16x32) per CDNA5 ISA 7.12.2: lane half h holds K in two
// contiguous groups of 8: halves [8h, 8h+8) (elements 0..7) and
// [16+8h, 16+8h+8) (elements 8..15).  -> two 16B vector loads per lane.
__device__ __forceinline__ int kOf(int e, int half) {
    int v = e >> 1, p = e & 1;
    return (v < 4) ? (half * 8 + v * 2 + p) : (16 + half * 8 + (v - 4) * 2 + p);
}

__device__ __forceinline__ v8f wmma_f16(v16h a, v16h b, v8f c) {
    return __builtin_amdgcn_wmma_f32_16x16x32_f16(false, a, false, b, (short)0, c, false, false);
}

__device__ __forceinline__ v16h cat8(v8h lo, v8h hi) {
    return __builtin_shufflevector(lo, hi, 0, 1, 2, 3, 4, 5, 6, 7,
                                           8, 9, 10, 11, 12, 13, 14, 15);
}

// A fragment from contiguous f16 row (halves [8h,8h+8) and [16+8h,16+8h+8))
__device__ __forceinline__ v16h loadAfragH(const _Float16* row, int half) {
    v8h lo = *(const v8h*)(row + 8 * half);
    v8h hi = *(const v8h*)(row + 16 + 8 * half);
    return cat8(lo, hi);
}

// A fragment from contiguous f32 row: two aligned v8f loads + convert
__device__ __forceinline__ v16h loadAfragF(const float* row, int half) {
    v8f f0 = *(const v8f*)(row + 8 * half);
    v8f f1 = *(const v8f*)(row + 16 + 8 * half);
    v16h a;
#pragma unroll
    for (int e = 0; e < 8; ++e) { a[e] = (_Float16)f0[e]; a[8 + e] = (_Float16)f1[e]; }
    return a;
}

// ---------------------------------------------------------------------------
// Kernel 0: transpose + convert the 4 weight matrices: Wt[k][n] = f16(W[n][k])
// grid = (768*768/256, 4).  Writes coalesced along n.
// ---------------------------------------------------------------------------
__global__ __launch_bounds__(256) void wtrans_kernel(
    const float* __restrict__ Wq, const float* __restrict__ Wk,
    const float* __restrict__ Wv, const float* __restrict__ Wo,
    _Float16* __restrict__ Wt)
{
    const int m = blockIdx.y;
    const float* W = (m == 0) ? Wq : (m == 1) ? Wk : (m == 2) ? Wv : Wo;
    _Float16* dst = Wt + (size_t)m * DMODEL * DMODEL;
    const int tid = blockIdx.x * 256 + threadIdx.x;
    const int n = tid % DMODEL, kk = tid / DMODEL;
    dst[(size_t)kk * DMODEL + n] = (_Float16)W[(size_t)n * DMODEL + kk];
}

// ---------------------------------------------------------------------------
// Kernel 1: fused Q/K/V projection.  One wave -> 16x64 output tile (4 WMMAs/k).
// Q,V written as f16 [seq][768]; K written transposed per head: Kt[b][h][d][s].
// grid.x = (512*12)/8, grid.y = 0/1/2 selects q/k/v.
// ---------------------------------------------------------------------------
__global__ __launch_bounds__(256) void proj_qkv_kernel(
    const float* __restrict__ q, const float* __restrict__ k, const float* __restrict__ v,
    const float* __restrict__ bq, const float* __restrict__ bk, const float* __restrict__ bv,
    const _Float16* __restrict__ Wt,
    _Float16* __restrict__ Qh, _Float16* __restrict__ Kt, _Float16* __restrict__ Vh)
{
    const int lane = threadIdx.x & 31;
    const int wave = threadIdx.x >> 5;
    const int wid  = blockIdx.x * 8 + wave;          // [0, 512*12)
    const int rBase = (wid / 12) * 16;
    const int cBase = (wid % 12) * 64;
    const int mrow = lane & 15;
    const int half = lane >> 4;

    const float* X; const float* bias; const _Float16* Wm;
    if (blockIdx.y == 0)      { X = q; bias = bq; Wm = Wt; }
    else if (blockIdx.y == 1) { X = k; bias = bk; Wm = Wt + (size_t)DMODEL * DMODEL; }
    else                      { X = v; bias = bv; Wm = Wt + (size_t)2 * DMODEL * DMODEL; }

    v8f acc[4];
#pragma unroll
    for (int t = 0; t < 4; ++t) acc[t] = (v8f){};

    for (int kt = 0; kt < DMODEL / 32; ++kt) {
        v16h a = loadAfragF(X + (size_t)(rBase + mrow) * DMODEL + kt * 32, half);
        const _Float16* wrow = Wm + (size_t)(kt * 32) * DMODEL + cBase + mrow;
#pragma unroll
        for (int t = 0; t < 4; ++t) {
            v16h b;
#pragma unroll
            for (int e = 0; e < 16; ++e) b[e] = wrow[(size_t)kOf(e, half) * DMODEL + t * 16];
            acc[t] = wmma_f16(a, b, acc[t]);
        }
    }

    if (blockIdx.y == 1) {
        // K: store transposed per head.  Lane has fixed out-col c -> fixed (h,d);
        // r indexes 8 consecutive seq positions -> one aligned v8h store.
        const int seq0 = rBase + 8 * half;
        const int bb = seq0 >> 9, s0 = seq0 & 511;
#pragma unroll
        for (int t = 0; t < 4; ++t) {
            const int c = cBase + t * 16 + mrow;
            const int h = c >> 6, d = c & 63;
            const float bi = bias[c];
            v8h pk;
#pragma unroll
            for (int r = 0; r < 8; ++r) pk[r] = (_Float16)(acc[t][r] + bi);
            *(v8h*)(Kt + ((size_t)(bb * NHEAD + h) * DK + d) * SLEN + s0) = pk;
        }
    } else {
        _Float16* Y = (blockIdx.y == 0) ? Qh : Vh;
#pragma unroll
        for (int t = 0; t < 4; ++t) {
            const float bi = bias[cBase + t * 16 + mrow];
#pragma unroll
            for (int r = 0; r < 8; ++r) {
                int m = r + 8 * half;
                Y[(size_t)(rBase + m) * DMODEL + cBase + t * 16 + mrow] =
                    (_Float16)(acc[t][r] + bi);
            }
        }
    }
}

// ---------------------------------------------------------------------------
// Kernel 2: flash-style attention.  One wave -> (batch, head, 16-query tile).
// Score B-fragments are 16 strided f16 loads from Kt; V fragments 16 strided f16
// loads from Vh; P re-layout D->A through per-wave LDS tile.
// ---------------------------------------------------------------------------
__global__ __launch_bounds__(256) void attn_kernel(
    const _Float16* __restrict__ Qh, const _Float16* __restrict__ Kt,
    const _Float16* __restrict__ Vh,
    const float* __restrict__ mask, const float* __restrict__ adj,
    _Float16* __restrict__ X2)
{
    __shared__ _Float16 plds[8 * 16 * 32];           // 1KB per wave
    const int lane = threadIdx.x & 31;
    const int wave = threadIdx.x >> 5;
    const int wid  = blockIdx.x * 8 + wave;          // [0, 192*32)
    const int qt   = wid & 31;
    const int bh   = wid >> 5;                       // b*12 + h
    const int h    = bh % NHEAD;
    const int bb   = bh / NHEAD;

    const int qBase = qt * 16;
    const int mrow  = lane & 15;
    const int half  = lane >> 4;
    const size_t headOff = (size_t)h * DK;
    const size_t seqBase = (size_t)bb * SLEN;
    const _Float16* KtH = Kt + (size_t)bh * DK * SLEN;

    _Float16* pbuf = plds + wave * (16 * 32);

    // Q A-fragments, two K=32 steps over d = 0..63 (2x 16B loads each)
    v16h qa[2];
    {
        const _Float16* qrow = Qh + (seqBase + qBase + mrow) * DMODEL + headOff;
        qa[0] = loadAfragH(qrow, half);
        qa[1] = loadAfragH(qrow + 32, half);
    }

    float mstat[8], lstat[8];
    v8f o[4];
#pragma unroll
    for (int r = 0; r < 8; ++r) { mstat[r] = -3.0e38f; lstat[r] = 0.0f; }
#pragma unroll
    for (int dt = 0; dt < 4; ++dt) o[dt] = (v8f){};

    for (int kt = 0; kt < SLEN / 32; ++kt) {
        const int keyBase = kt * 32;

        // scores for two 16-key tiles; B[d][key]: lane = key, element = d (table order)
        v8f sc[2];
#pragma unroll
        for (int j = 0; j < 2; ++j) {
            v8f z = {};
#pragma unroll
            for (int s = 0; s < 2; ++s) {
                v16h b;
#pragma unroll
                for (int e = 0; e < 16; ++e)
                    b[e] = KtH[(size_t)(s * 32 + kOf(e, half)) * SLEN + keyBase + j * 16 + mrow];
                z = wmma_f16(qa[s], b, z);
            }
            sc[j] = z;
        }

        // scale + padding mask + adjacency bias
        float mk0 = mask[seqBase + keyBase + mrow] * NEG_BIG;
        float mk1 = mask[seqBase + keyBase + 16 + mrow] * NEG_BIG;
        const float* adjb = adj + ((size_t)bb * SLEN + qBase) * SLEN + keyBase;
#pragma unroll
        for (int r = 0; r < 8; ++r) {
            size_t qrow = (size_t)(r + 8 * half) * SLEN;
            sc[0][r] = sc[0][r] * 0.125f + mk0 + adjb[qrow + mrow];
            sc[1][r] = sc[1][r] * 0.125f + mk1 + adjb[qrow + 16 + mrow];
        }

        __builtin_amdgcn_wave_barrier();             // order vs prev iter's LDS reads

        // online softmax per query row; stash P to LDS (row-major 16x32)
#pragma unroll
        for (int r = 0; r < 8; ++r) {
            float t = fmaxf(sc[0][r], sc[1][r]);
#pragma unroll
            for (int off = 1; off < 16; off <<= 1)
                t = fmaxf(t, __shfl_xor(t, off, 32));
            float mnew = fmaxf(mstat[r], t);
            float corr = __expf(mstat[r] - mnew);
            float p0 = __expf(sc[0][r] - mnew);
            float p1 = __expf(sc[1][r] - mnew);
            float rs = p0 + p1;
#pragma unroll
            for (int off = 1; off < 16; off <<= 1)
                rs += __shfl_xor(rs, off, 32);
            lstat[r] = lstat[r] * corr + rs;
            mstat[r] = mnew;
#pragma unroll
            for (int dt = 0; dt < 4; ++dt) o[dt][r] *= corr;
            pbuf[(r + 8 * half) * 32 + mrow]      = (_Float16)p0;
            pbuf[(r + 8 * half) * 32 + 16 + mrow] = (_Float16)p1;
        }

        __builtin_amdgcn_wave_barrier();             // stores before cross-lane reads

        // P as A-fragment: two aligned 16B LDS loads
        v16h pa = loadAfragH(pbuf + mrow * 32, half);

        // PV: V tile 32 keys x 64 dims; lane = dim column, element = key (table order)
#pragma unroll
        for (int dt = 0; dt < 4; ++dt) {
            v16h vb;
#pragma unroll
            for (int e = 0; e < 16; ++e)
                vb[e] = Vh[(seqBase + keyBase + kOf(e, half)) * DMODEL + headOff + dt * 16 + mrow];
            o[dt] = wmma_f16(pa, vb, o[dt]);
        }
    }

    // normalize and store head output into [8192 x 768] f16 workspace
#pragma unroll
    for (int r = 0; r < 8; ++r) {
        float inv = 1.0f / lstat[r];
        int m = r + 8 * half;
#pragma unroll
        for (int dt = 0; dt < 4; ++dt)
            X2[(seqBase + qBase + m) * DMODEL + headOff + dt * 16 + mrow] =
                (_Float16)(o[dt][r] * inv);
    }
}

// ---------------------------------------------------------------------------
// Kernel 3: output projection.  out = fp32(X2 @ Wo^T + bo); 16x64 tile per wave.
// ---------------------------------------------------------------------------
__global__ __launch_bounds__(256) void proj_out_kernel(
    const _Float16* __restrict__ X2, const _Float16* __restrict__ WtO,
    const float* __restrict__ bo, float* __restrict__ out)
{
    const int lane = threadIdx.x & 31;
    const int wave = threadIdx.x >> 5;
    const int wid  = blockIdx.x * 8 + wave;
    const int rBase = (wid / 12) * 16;
    const int cBase = (wid % 12) * 64;
    const int mrow = lane & 15;
    const int half = lane >> 4;

    v8f acc[4];
#pragma unroll
    for (int t = 0; t < 4; ++t) acc[t] = (v8f){};

    for (int kt = 0; kt < DMODEL / 32; ++kt) {
        v16h a = loadAfragH(X2 + (size_t)(rBase + mrow) * DMODEL + kt * 32, half);
        const _Float16* wrow = WtO + (size_t)(kt * 32) * DMODEL + cBase + mrow;
#pragma unroll
        for (int t = 0; t < 4; ++t) {
            v16h b;
#pragma unroll
            for (int e = 0; e < 16; ++e) b[e] = wrow[(size_t)kOf(e, half) * DMODEL + t * 16];
            acc[t] = wmma_f16(a, b, acc[t]);
        }
    }
#pragma unroll
    for (int t = 0; t < 4; ++t) {
        const float bi = bo[cBase + t * 16 + mrow];
#pragma unroll
        for (int r = 0; r < 8; ++r) {
            int m = r + 8 * half;
            out[(size_t)(rBase + m) * DMODEL + cBase + t * 16 + mrow] = acc[t][r] + bi;
        }
    }
}

extern "C" void kernel_launch(void* const* d_in, const int* in_sizes, int n_in,
                              void* d_out, int out_size, void* d_ws, size_t ws_size,
                              hipStream_t stream) {
    const float* q    = (const float*)d_in[0];
    const float* k    = (const float*)d_in[1];
    const float* v    = (const float*)d_in[2];
    const float* mask = (const float*)d_in[3];
    const float* adj  = (const float*)d_in[4];
    const float* Wq   = (const float*)d_in[5];
    const float* bq   = (const float*)d_in[6];
    const float* Wk   = (const float*)d_in[7];
    const float* bk   = (const float*)d_in[8];
    const float* Wv   = (const float*)d_in[9];
    const float* bv   = (const float*)d_in[10];
    const float* Wo   = (const float*)d_in[11];
    const float* bo   = (const float*)d_in[12];
    float* out = (float*)d_out;

    const size_t wsz = (size_t)DMODEL * DMODEL;      // 589824 halves per weight
    const size_t n   = (size_t)BATCH * SLEN * DMODEL;// 6291456 halves per tensor
    _Float16* ws = (_Float16*)d_ws;
    _Float16* Wt = ws;                               // 4 transposed f16 weights
    _Float16* Qh = ws + 4 * wsz;
    _Float16* Kt = Qh + n;                           // [b][h][d][s]
    _Float16* Vh = Kt + n;
    _Float16* X2 = Vh + n;

    dim3 gW((DMODEL * DMODEL) / 256, 4, 1);
    wtrans_kernel<<<gW, 256, 0, stream>>>(Wq, Wk, Wv, Wo, Wt);

    dim3 gProj((512 * 12) / 8, 3, 1);                // 16x64 tile per wave
    proj_qkv_kernel<<<gProj, 256, 0, stream>>>(q, k, v, bq, bk, bv, Wt, Qh, Kt, Vh);

    dim3 gAttn((BATCH * NHEAD * (SLEN / 16)) / 8, 1, 1);
    attn_kernel<<<gAttn, 256, 0, stream>>>(Qh, Kt, Vh, mask, adj, X2);

    dim3 gOut((512 * 12) / 8, 1, 1);
    proj_out_kernel<<<gOut, 256, 0, stream>>>(X2, Wt + 3 * wsz, bo, out);
}
